// ProgressiveBExampleLM_5875515261424
// MI455X (gfx1250) — compile-verified, matrix operand order checked by reading, not run on hardware
//
#include <hip/hip_runtime.h>
#include <hip/hip_bf16.h>

#define BATCH  2
#define SEQ    2048
#define DIM    512
#define VOCAB  32000
#define WINDOW 4
#define TOPK   4

typedef __attribute__((ext_vector_type(16))) _Float16 v16h;
typedef __attribute__((ext_vector_type(8)))  _Float16 v8h;
typedef __attribute__((ext_vector_type(8)))  float    v8f;
typedef int v4i __attribute__((vector_size(16)));   // matches builtin signature

#if defined(__HIP_DEVICE_COMPILE__) && __has_builtin(__builtin_amdgcn_global_load_async_to_lds_b128) && __has_builtin(__builtin_amdgcn_s_wait_asynccnt)
#define ASYNC_LDS 1
#else
#define ASYNC_LDS 0
#endif

static __device__ __forceinline__ float wave_sum32(float v) {
#pragma unroll
  for (int off = 16; off > 0; off >>= 1) v += __shfl_xor(v, off, 32);
  return v;
}

static __device__ __forceinline__ v16h mk16(v8h lo, v8h hi) {
  return __builtin_shufflevector(lo, hi, 0,1,2,3,4,5,6,7,8,9,10,11,12,13,14,15);
}

static __device__ __forceinline__ v8h zero8h() {
  v8h r;
#pragma unroll
  for (int i = 0; i < 8; i++) r[i] = (_Float16)0.f;
  return r;
}

// 16-byte global -> LDS copy: async direct-to-LDS when available, else bounce.
static __device__ __forceinline__ void cp16(const _Float16* g, _Float16* l) {
#if ASYNC_LDS
  __builtin_amdgcn_global_load_async_to_lds_b128(
      (__attribute__((address_space(1))) v4i*)g,
      (__attribute__((address_space(3))) v4i*)l, 0, 0);
#else
  *(v8h*)l = *(const v8h*)g;
#endif
}

static __device__ __forceinline__ void async_wait0() {
#if ASYNC_LDS
  __builtin_amdgcn_s_wait_asynccnt(0);
#endif
}

// ---------------------------------------------------------------- embedding
__global__ void embed_gather(const int* __restrict__ tok, const float* __restrict__ emb,
                             float* __restrict__ st, float* __restrict__ vl) {
  long row = blockIdx.x;                       // [0, BATCH*SEQ)
  int t = tok[row];
  const float* e = emb + (long)t * DIM;
  float* s = st + row * DIM;
  float* v = vl + row * DIM;
  for (int d = threadIdx.x; d < DIM; d += blockDim.x) {
    float x = e[d]; s[d] = x; v[d] = x;
  }
}

// broadcast positional tables into both batch slots of state & val
__global__ void bcast_pe(const float* __restrict__ pe, float* __restrict__ st,
                         float* __restrict__ vl, long n) {
  long i = (long)blockIdx.x * blockDim.x + threadIdx.x;
  if (i >= n) return;
  float v = pe[i];
  st[i] = v; st[n + i] = v;
  vl[i] = v; vl[n + i] = v;
}

// ------------------------------------------------- causal window propagation
__global__ __launch_bounds__(256)
void window_prop(const float* __restrict__ st, const float* __restrict__ vl,
                 const float* __restrict__ w,
                 float* __restrict__ nst, float* __restrict__ nvl) {
  int wave = threadIdx.x >> 5, lane = threadIdx.x & 31;
  long row = (long)blockIdx.x * 8 + wave;
  long b = row / SEQ;
  int  s = (int)(row % SEQ);
  const float* sb = st + b * (long)SEQ * DIM;
  const float* vb = vl + b * (long)SEQ * DIM;
  float self_s[16], self_v[16], wl[16];
#pragma unroll
  for (int i = 0; i < 16; i++) {
    int d = lane + 32 * i;
    self_s[i] = sb[(long)s * DIM + d];
    self_v[i] = vb[(long)s * DIM + d];
    wl[i] = w[d];
  }
  float sc[WINDOW];
#pragma unroll
  for (int j = 1; j <= WINDOW; j++) {
    float p = 0.f;
    if (s - j >= 0) {
      const float* nb = sb + (long)(s - j) * DIM;
#pragma unroll
      for (int i = 0; i < 16; i++) p += self_s[i] * wl[i] * nb[lane + 32 * i];
    }
    p = wave_sum32(p);
    sc[j - 1] = (s - j >= 0) ? p : -1e9f;
  }
  float mx = sc[0];
#pragma unroll
  for (int j = 1; j < WINDOW; j++) mx = fmaxf(mx, sc[j]);
  float pr[WINDOW]; float sum = 0.f;
#pragma unroll
  for (int j = 0; j < WINDOW; j++) { pr[j] = expf(sc[j] - mx); sum += pr[j]; }
  float inv = 1.f / sum;
#pragma unroll
  for (int j = 0; j < WINDOW; j++) pr[j] = (s - (j + 1) >= 0) ? pr[j] * inv : 0.f;
  float ds[16], dv[16];
#pragma unroll
  for (int i = 0; i < 16; i++) { ds[i] = 0.f; dv[i] = 0.f; }
#pragma unroll
  for (int j = 0; j < WINDOW; j++) {
    if (s - (j + 1) >= 0) {
      const float* nb = sb + (long)(s - 1 - j) * DIM;
      const float* nv = vb + (long)(s - 1 - j) * DIM;
#pragma unroll
      for (int i = 0; i < 16; i++) {
        int d = lane + 32 * i;
        ds[i] += pr[j] * nb[d];
        dv[i] += pr[j] * nv[d];
      }
    }
  }
  float* os = nst + row * DIM;
  float* ov = nvl + row * DIM;
#pragma unroll
  for (int i = 0; i < 16; i++) {
    int d = lane + 32 * i;
    os[d] = self_s[i] + 0.25f * ds[i];   // S_DELTA
    ov[d] = self_v[i] + 0.25f * dv[i];
  }
}

// ------------------------------------------------------------- small utility
__global__ void scale_to_h(const float* __restrict__ x, const float* __restrict__ w,
                           _Float16* __restrict__ o, long n) {
  long i = (long)blockIdx.x * 256 + threadIdx.x;
  if (i >= n) return;
  o[i] = (_Float16)(x[i] * w[i & (DIM - 1)]);
}

__global__ void to_h(const float* __restrict__ x, _Float16* __restrict__ o, long n) {
  long i = (long)blockIdx.x * 256 + threadIdx.x;
  if (i >= n) return;
  o[i] = (_Float16)x[i];
}

__global__ void axpy_k(float* __restrict__ y, const float* __restrict__ x, float c, long n) {
  long i = (long)blockIdx.x * 256 + threadIdx.x;
  if (i >= n) return;
  y[i] += c * x[i];
}

__global__ void tanh_k(float* __restrict__ x, long n) {
  long i = (long)blockIdx.x * 256 + threadIdx.x;
  if (i >= n) return;
  x[i] = tanhf(x[i]);
}

__global__ __launch_bounds__(256)
void layernorm_k(float* __restrict__ x, const float* __restrict__ g,
                 const float* __restrict__ bb, long nrows) {
  int wave = threadIdx.x >> 5, lane = threadIdx.x & 31;
  long row = (long)blockIdx.x * 8 + wave;
  if (row >= nrows) return;
  float* p = x + row * DIM;
  float v[16]; float s = 0.f;
#pragma unroll
  for (int i = 0; i < 16; i++) { v[i] = p[lane + 32 * i]; s += v[i]; }
  float mu = wave_sum32(s) * (1.f / DIM);
  float q = 0.f;
#pragma unroll
  for (int i = 0; i < 16; i++) { float d = v[i] - mu; q += d * d; }
  float var = wave_sum32(q) * (1.f / DIM);
  float r = rsqrtf(var + 1e-5f);
#pragma unroll
  for (int i = 0; i < 16; i++) {
    int d = lane + 32 * i;
    p[d] = (v[i] - mu) * r * g[d] + bb[d];
  }
}

// ---------------------------------------------- fused score-GEMM + top-4 route
// WMMA scores (f16 in / f32 acc); src tiles streamed via async global->LDS with
// double buffering so the fetch of tile t+1 overlaps the WMMAs of tile t.
__global__ __launch_bounds__(64)
void topk_attend(const _Float16* __restrict__ hA,   // [B,Ndst,DIM] = f16(dst_state * w)
                 const _Float16* __restrict__ hS,   // [B,Nsrc,DIM] = f16(src_state)
                 const float* __restrict__ srcS, const float* __restrict__ srcV,
                 float* __restrict__ dS, float* __restrict__ dV,
                 int Ndst, int Nsrc) {
  __shared__ _Float16 ldsA[2][16 * DIM];   // per-wave dst tile    (2 x 16KB)
  __shared__ _Float16 ldsB[2][16 * DIM];   // double-buffered src  (2 x 16KB)
  __shared__ float    sc[2][16][16];
  __shared__ float    pbuf[2][16][TOPK];
  __shared__ int      ibuf[2][16][TOPK];

  int wave = threadIdx.x >> 5, lane = threadIdx.x & 31;
  int b  = blockIdx.y;
  int m0 = (blockIdx.x * 2 + wave) * 16;

  // stage own A tile (zero-pad rows beyond Ndst)
  const _Float16* Ab = hA + (long)b * Ndst * DIM;
  for (int r = 0; r < 16; r++) {
    int m = m0 + r;
    v8h a0, a1;
    if (m < Ndst) {
      const v8h* sp = (const v8h*)(Ab + (long)m * DIM + lane * 16);
      a0 = sp[0]; a1 = sp[1];
    } else {
      a0 = zero8h(); a1 = a0;
    }
    *(v8h*)&ldsA[wave][r * DIM + lane * 16]     = a0;
    *(v8h*)&ldsA[wave][r * DIM + lane * 16 + 8] = a1;
  }

  float tv0 = -3e38f, tv1 = -3e38f, tv2 = -3e38f, tv3 = -3e38f;
  int   ti0 = 0, ti1 = 0, ti2 = 0, ti3 = 0;

  const _Float16* Bb = hS + (long)b * Nsrc * DIM;
  int ntile = (Nsrc + 15) >> 4;
  int arow  = lane & 15;
  int koff  = (lane >> 4) * 8;     // ISA 16-bit frag K split: {0..7,16..23}/{8..15,24..31}
  const int slice = threadIdx.x * 128;   // 64 threads x 256B cover one 16KB tile

  {   // prologue: issue tile 0 into buffer 0
    const _Float16* g = Bb + slice;
    _Float16* l = &ldsB[0][slice];
#pragma unroll
    for (int i = 0; i < 16; i++) cp16(g + i * 8, l + i * 8);
  }

  for (int t = 0; t < ntile; t++) {
    int cur = t & 1;
    async_wait0();        // my part of tile t landed in LDS
    __syncthreads();      // everyone's part landed; prev compute on cur^1 done
    if (t + 1 < ntile) {  // prefetch tile t+1 while computing on tile t
      const _Float16* g = Bb + (long)(t + 1) * 16 * DIM + slice;
      _Float16* l = &ldsB[cur ^ 1][slice];
#pragma unroll
      for (int i = 0; i < 16; i++) cp16(g + i * 8, l + i * 8);
    }

    v8f acc = {};
    const _Float16* ap = &ldsA[wave][arow * DIM];
    const _Float16* bp = &ldsB[cur][arow * DIM];
#pragma unroll
    for (int k = 0; k < DIM; k += 32) {
      v16h af = mk16(*(const v8h*)(ap + k + koff), *(const v8h*)(ap + k + koff + 16));
      v16h bf = mk16(*(const v8h*)(bp + k + koff), *(const v8h*)(bp + k + koff + 16));
      acc = __builtin_amdgcn_wmma_f32_16x16x32_f16(false, af, false, bf,
                                                   (short)0, acc, false, false);
    }
    {   // dump 16x16 score tile: M = r2 + 8*(lane>>4), N = lane&15
      int nlo = lane & 15, mg = (lane >> 4) * 8;
#pragma unroll
      for (int r2 = 0; r2 < 8; r2++) sc[wave][mg + r2][nlo] = acc[r2];
    }
    asm volatile("s_wait_dscnt 0" ::: "memory");   // wave-local LDS RAW
    if (lane < 16) {
#pragma unroll
      for (int j = 0; j < 16; j++) {
        int ng = t * 16 + j;
        if (ng >= Nsrc) break;     // OOB src rows (garbage data) never scanned
        float v = sc[wave][lane][j];
        if (v > tv3) {
          if (v > tv0)      { tv3=tv2;ti3=ti2; tv2=tv1;ti2=ti1; tv1=tv0;ti1=ti0; tv0=v;ti0=ng; }
          else if (v > tv1) { tv3=tv2;ti3=ti2; tv2=tv1;ti2=ti1; tv1=v;ti1=ng; }
          else if (v > tv2) { tv3=tv2;ti3=ti2; tv2=v;ti2=ng; }
          else              { tv3=v;ti3=ng; }
        }
      }
    }
  }

  if (lane < 16) {   // softmax over top-4 (sorted descending -> tv0 is max)
    float e0 = expf(tv0 - tv0), e1 = expf(tv1 - tv0),
          e2 = expf(tv2 - tv0), e3 = expf(tv3 - tv0);
    float inv = 1.f / (e0 + e1 + e2 + e3);
    pbuf[wave][lane][0] = e0 * inv; pbuf[wave][lane][1] = e1 * inv;
    pbuf[wave][lane][2] = e2 * inv; pbuf[wave][lane][3] = e3 * inv;
    ibuf[wave][lane][0] = ti0; ibuf[wave][lane][1] = ti1;
    ibuf[wave][lane][2] = ti2; ibuf[wave][lane][3] = ti3;
  }
  asm volatile("s_wait_dscnt 0" ::: "memory");

  // cooperative coalesced gather + weighted sum in f32
  const float* Sb = srcS + (long)b * Nsrc * DIM;
  const float* Vb = srcV + (long)b * Nsrc * DIM;
  float* oS = dS + (long)b * Ndst * DIM;
  float* oV = dV + (long)b * Ndst * DIM;
  for (int r = 0; r < 16; r++) {
    int m = m0 + r;
    if (m >= Ndst) break;
    float p0 = pbuf[wave][r][0], p1 = pbuf[wave][r][1],
          p2 = pbuf[wave][r][2], p3 = pbuf[wave][r][3];
    long i0 = ibuf[wave][r][0], i1 = ibuf[wave][r][1],
         i2 = ibuf[wave][r][2], i3 = ibuf[wave][r][3];
    const float *s0 = Sb + i0 * DIM, *s1 = Sb + i1 * DIM, *s2 = Sb + i2 * DIM, *s3 = Sb + i3 * DIM;
    const float *v0 = Vb + i0 * DIM, *v1 = Vb + i1 * DIM, *v2 = Vb + i2 * DIM, *v3 = Vb + i3 * DIM;
#pragma unroll 4
    for (int i = 0; i < 16; i++) {
      int d = lane + 32 * i;
      oS[(long)m * DIM + d] = p0 * s0[d] + p1 * s1[d] + p2 * s2[d] + p3 * s3[d];
      oV[(long)m * DIM + d] = p0 * v0[d] + p1 * v1[d] + p2 * v2[d] + p3 * v3[d];
    }
  }
}

// ----------------------------------------------------------- readout GEMM
// [4096,512] x [512,32000] -> f32 logits. W K-chunks streamed async into a
// double-buffered LDS tile so the next chunk's fetch overlaps this chunk's WMMA.
__global__ __launch_bounds__(256)
void readout_gemm(const _Float16* __restrict__ A, const _Float16* __restrict__ W,
                  const float* __restrict__ bias, float* __restrict__ out) {
  __shared__ _Float16 ldsA[16 * DIM];       // 16KB (16 M-rows, full K)
  __shared__ _Float16 ldsW[2][32 * 128];    // 2 x 8KB (32 K-rows, 128 N-cols)
  int wave = threadIdx.x >> 5, lane = threadIdx.x & 31;
  int m0 = blockIdx.y * 16;
  int n0 = blockIdx.x * 128;
  {
    int r = threadIdx.x >> 4;
    int c = (threadIdx.x & 15) * 32;
    const v8h* sp = (const v8h*)(A + (long)(m0 + r) * DIM + c);
    v8h* dp = (v8h*)&ldsA[r * DIM + c];
    dp[0] = sp[0]; dp[1] = sp[1]; dp[2] = sp[2]; dp[3] = sp[3];
  }
  int wr = threadIdx.x >> 3;                 // 32 K-rows, 8 threads/row
  int wc = (threadIdx.x & 7) * 16;           // 16 halves (32B) per thread
  const _Float16* Wbase = W + (long)wr * VOCAB + n0 + wc;
  {   // prologue: K-chunk 0 into buffer 0
    _Float16* l = &ldsW[0][wr * 128 + wc];
    cp16(Wbase, l); cp16(Wbase + 8, l + 8);
  }
  v8f acc = {};
  int arow = lane & 15, koff = (lane >> 4) * 8;
  int ncol = wave * 16 + (lane & 15);
  for (int k0 = 0; k0 < DIM; k0 += 32) {
    int cur = (k0 >> 5) & 1;
    async_wait0();
    __syncthreads();
    if (k0 + 32 < DIM) {   // prefetch next K-chunk
      const _Float16* g = Wbase + (long)(k0 + 32) * VOCAB;
      _Float16* l = &ldsW[cur ^ 1][wr * 128 + wc];
      cp16(g, l); cp16(g + 8, l + 8);
    }
    v16h af = mk16(*(const v8h*)&ldsA[arow * DIM + k0 + koff],
                   *(const v8h*)&ldsA[arow * DIM + k0 + koff + 16]);
    v16h bf;
#pragma unroll
    for (int i = 0; i < 8; i++) {
      bf[i]     = ldsW[cur][(koff + i) * 128 + ncol];
      bf[i + 8] = ldsW[cur][(koff + 16 + i) * 128 + ncol];
    }
    acc = __builtin_amdgcn_wmma_f32_16x16x32_f16(false, af, false, bf,
                                                 (short)0, acc, false, false);
  }
  int n = n0 + ncol;
  float bv = bias[n];
  int mg = (lane >> 4) * 8;
#pragma unroll
  for (int r2 = 0; r2 < 8; r2++) {
    long m = m0 + mg + r2;
    out[m * (long)VOCAB + n] = acc[r2] + bv;
  }
}

// ================================================================== host
extern "C" void kernel_launch(void* const* d_in, const int* in_sizes, int n_in,
                              void* d_out, int out_size, void* d_ws, size_t ws_size,
                              hipStream_t stream) {
  (void)in_sizes; (void)n_in; (void)out_size; (void)ws_size;
  const int*   tokens    = (const int*)  d_in[0];
  const float* embed     = (const float*)d_in[1];
  const float* readout_w = (const float*)d_in[2];
  const float* readout_b = (const float*)d_in[3];
  const float* w_s   = (const float*)d_in[4];
  const float* w_se  = (const float*)d_in[5];
  const float* w_sc  = (const float*)d_in[6];
  const float* w_s2e = (const float*)d_in[7];
  const float* w_e2s = (const float*)d_in[8];
  const float* w_s2c = (const float*)d_in[9];
  const float* w_c2s = (const float*)d_in[10];
  const float* ln_s_g = (const float*)d_in[11]; const float* ln_s_b = (const float*)d_in[12];
  const float* ln_e_g = (const float*)d_in[13]; const float* ln_e_b = (const float*)d_in[14];
  const float* ln_c_g = (const float*)d_in[15]; const float* ln_c_b = (const float*)d_in[16];
  const float* pe_e[3] = {(const float*)d_in[17], (const float*)d_in[19], (const float*)d_in[21]};
  const float* pe_c[3] = {(const float*)d_in[18], (const float*)d_in[20], (const float*)d_in[22]};

  const int NEXP[3] = {2151, 2253, 2458};
  const int NCMP[3] = {1844, 1639, 1434};
  const int LAY[3]  = {2, 2, 1};
  const float AL[3][3] = {{0.3f, 0.25f, 0.15f}, {0.65f, 0.55f, 0.35f}, {1.0f, 0.9f, 0.8f}};
  const int NMAX = 2458;

  char* p = (char*)d_ws;
  auto alloc = [&](size_t bytes) -> char* {
    char* r = p; p += (bytes + 255) & ~(size_t)255; return r;
  };
  size_t seqB = (size_t)BATCH * SEQ * DIM * 4;
  size_t bigB = (size_t)BATCH * NMAX * DIM * 4;
  size_t cmpB = (size_t)BATCH * 1844 * DIM * 4;
  float* s_state = (float*)alloc(seqB); float* s_val = (float*)alloc(seqB);
  float* s_ts    = (float*)alloc(seqB); float* s_tv  = (float*)alloc(seqB);
  float* e_state = (float*)alloc(bigB); float* e_val = (float*)alloc(bigB);
  float* c_state = (float*)alloc(cmpB); float* c_val = (float*)alloc(cmpB);
  float* d_state = (float*)alloc(bigB); float* d_val = (float*)alloc(bigB);
  _Float16* hA  = (_Float16*)alloc((size_t)BATCH * NMAX * DIM * 2);
  _Float16* hA2 = (_Float16*)alloc((size_t)BATCH * SEQ  * DIM * 2);
  _Float16* hB  = (_Float16*)alloc((size_t)BATCH * NMAX * DIM * 2);
  _Float16* hW  = (_Float16*)alloc((size_t)DIM * VOCAB * 2);

  auto blocks = [](long n) -> unsigned { return (unsigned)((n + 255) / 256); };
  long nS = (long)BATCH * SEQ * DIM;
  long nW = (long)DIM * VOCAB;

  to_h<<<blocks(nW), 256, 0, stream>>>(readout_w, hW, nW);
  embed_gather<<<BATCH * SEQ, 128, 0, stream>>>(tokens, embed, s_state, s_val);

  int li = 0;
  for (int st = 0; st < 3; st++) {
    int Ne = NEXP[st], Nc = NCMP[st];
    long nE = (long)BATCH * Ne * DIM, nC = (long)BATCH * Nc * DIM;
    bcast_pe<<<blocks((long)Ne * DIM), 256, 0, stream>>>(pe_e[st], e_state, e_val, (long)Ne * DIM);
    bcast_pe<<<blocks((long)Nc * DIM), 256, 0, stream>>>(pe_c[st], c_state, c_val, (long)Nc * DIM);
    float alpha = AL[st][0], bs2b = AL[st][1], bb2s = AL[st][2];
    float cx = 0.15f * bs2b * alpha;   // X_DELTA * beta_s_to_b * alpha
    float cb = 0.2f * alpha;           // B_DELTA * alpha
    float cs = 0.15f * bb2s;           // X_DELTA * beta_b_to_s
    for (int l = 0; l < LAY[st]; l++) {
      // 1) causal window propagation on S (out-of-place, then pointer swap)
      window_prop<<<BATCH * SEQ / 8, 256, 0, stream>>>(s_state, s_val, w_s + li * DIM, s_ts, s_tv);
      { float* t = s_state; s_state = s_ts; s_ts = t; t = s_val; s_val = s_tv; s_tv = t; }

      // 2) E <- S
      scale_to_h<<<blocks(nE), 256, 0, stream>>>(e_state, w_s2e + li * DIM, hA, nE);
      to_h<<<blocks(nS), 256, 0, stream>>>(s_state, hB, nS);
      topk_attend<<<dim3((Ne + 31) / 32, BATCH), 64, 0, stream>>>(hA, hB, s_state, s_val,
                                                                  d_state, d_val, Ne, SEQ);
      axpy_k<<<blocks(nE), 256, 0, stream>>>(e_state, d_state, cx, nE);
      axpy_k<<<blocks(nE), 256, 0, stream>>>(e_val, d_val, cx, nE);

      // 3) C <- S (hB still holds f16 s_state)
      scale_to_h<<<blocks(nC), 256, 0, stream>>>(c_state, w_s2c + li * DIM, hA, nC);
      topk_attend<<<dim3((Nc + 31) / 32, BATCH), 64, 0, stream>>>(hA, hB, s_state, s_val,
                                                                  d_state, d_val, Nc, SEQ);
      axpy_k<<<blocks(nC), 256, 0, stream>>>(c_state, d_state, cx, nC);
      axpy_k<<<blocks(nC), 256, 0, stream>>>(c_val, d_val, cx, nC);

      // 4) E self
      scale_to_h<<<blocks(nE), 256, 0, stream>>>(e_state, w_se + li * DIM, hA, nE);
      to_h<<<blocks(nE), 256, 0, stream>>>(e_state, hB, nE);
      topk_attend<<<dim3((Ne + 31) / 32, BATCH), 64, 0, stream>>>(hA, hB, e_state, e_val,
                                                                  d_state, d_val, Ne, Ne);
      axpy_k<<<blocks(nE), 256, 0, stream>>>(e_state, d_state, cb, nE);
      axpy_k<<<blocks(nE), 256, 0, stream>>>(e_val, d_val, cb, nE);

      // 5) C self
      scale_to_h<<<blocks(nC), 256, 0, stream>>>(c_state, w_sc + li * DIM, hA, nC);
      to_h<<<blocks(nC), 256, 0, stream>>>(c_state, hB, nC);
      topk_attend<<<dim3((Nc + 31) / 32, BATCH), 64, 0, stream>>>(hA, hB, c_state, c_val,
                                                                  d_state, d_val, Nc, Nc);
      axpy_k<<<blocks(nC), 256, 0, stream>>>(c_state, d_state, cb, nC);
      axpy_k<<<blocks(nC), 256, 0, stream>>>(c_val, d_val, cb, nC);

      // 6/7) S <- E and S <- C (both A-operands built from pre-update s_state)
      scale_to_h<<<blocks(nS), 256, 0, stream>>>(s_state, w_e2s + li * DIM, hA, nS);
      scale_to_h<<<blocks(nS), 256, 0, stream>>>(s_state, w_c2s + li * DIM, hA2, nS);
      to_h<<<blocks(nE), 256, 0, stream>>>(e_state, hB, nE);
      topk_attend<<<dim3(SEQ / 32, BATCH), 64, 0, stream>>>(hA, hB, e_state, e_val,
                                                            d_state, d_val, SEQ, Ne);
      axpy_k<<<blocks(nS), 256, 0, stream>>>(s_state, d_state, cs, nS);
      axpy_k<<<blocks(nS), 256, 0, stream>>>(s_val, d_val, cs, nS);
      to_h<<<blocks(nC), 256, 0, stream>>>(c_state, hB, nC);
      topk_attend<<<dim3(SEQ / 32, BATCH), 64, 0, stream>>>(hA2, hB, c_state, c_val,
                                                            d_state, d_val, SEQ, Nc);
      axpy_k<<<blocks(nS), 256, 0, stream>>>(s_state, d_state, cs, nS);
      axpy_k<<<blocks(nS), 256, 0, stream>>>(s_val, d_val, cs, nS);

      // 8) stabilize: tanh(state), layernorm(val)
      tanh_k<<<blocks(nS), 256, 0, stream>>>(s_state, nS);
      layernorm_k<<<(unsigned)((BATCH * SEQ + 7) / 8), 256, 0, stream>>>(
          s_val, ln_s_g + li * DIM, ln_s_b + li * DIM, (long)BATCH * SEQ);
      tanh_k<<<blocks(nE), 256, 0, stream>>>(e_state, nE);
      layernorm_k<<<(unsigned)(((long)BATCH * Ne + 7) / 8), 256, 0, stream>>>(
          e_val, ln_e_g + li * DIM, ln_e_b + li * DIM, (long)BATCH * Ne);
      tanh_k<<<blocks(nC), 256, 0, stream>>>(c_state, nC);
      layernorm_k<<<(unsigned)(((long)BATCH * Nc + 7) / 8), 256, 0, stream>>>(
          c_val, ln_c_g + li * DIM, ln_c_b + li * DIM, (long)BATCH * Nc);
      li++;
    }
  }

  // readout: logits = s_val @ W + b  (WMMA f16 GEMM)
  to_h<<<blocks(nS), 256, 0, stream>>>(s_val, hB, nS);
  readout_gemm<<<dim3(VOCAB / 128, BATCH * SEQ / 16), 256, 0, stream>>>(
      hB, hW, readout_b, (float*)d_out);
}
